// PolygonPredictionLayer_18167711662402
// MI455X (gfx1250) — matrix operand order repaired and checked by reading, not compiled, not run on hardware
//
#include <hip/hip_runtime.h>

// ---------------------------------------------------------------------------
// PolygonPredictionLayer for MI455X (gfx1250, wave32, WMMA).
// bf16 data / f32 accumulate via v_wmma_f32_16x16x32_bf16.
// GEMM tile staging uses CDNA5 async global->LDS DMA (ASYNCcnt) + double buffer.
// ---------------------------------------------------------------------------

typedef __attribute__((ext_vector_type(16))) __bf16 v16bf;
typedef __attribute__((ext_vector_type(8)))  float  v8f;

__device__ __forceinline__ v8f wmma_bf16(v16bf a, v16bf b, v8f c) {
    return __builtin_amdgcn_wmma_f32_16x16x32_bf16(
        /*neg_a=*/false, a, /*neg_b=*/false, b,
        /*c_mod=*/(short)0, c, /*reuse_a=*/false, /*reuse_b=*/false);
}

// Async DMA of 16 bytes global -> LDS (no VGPR round trip, ASYNCcnt-tracked).
__device__ __forceinline__ void async_ld_b128(const void* gptr, void* lptr) {
    unsigned long long ga = (unsigned long long)(size_t)gptr;
    unsigned int       la = (unsigned int)(size_t)lptr;   // low 32b of generic = LDS offset
    asm volatile("global_load_async_to_lds_b128 %0, %1, off" : : "v"(la), "v"(ga) : "memory");
}

// ---------------------------------------------------------------------------
// Elementwise f32 -> bf16 convert (weight prep and chaining)
// ---------------------------------------------------------------------------
__global__ void cvt_f32_bf16(const float* __restrict__ x, __bf16* __restrict__ y, int n) {
    int i = blockIdx.x * 256 + threadIdx.x;
    if (i < n) y[i] = (__bf16)x[i];
}

// query_features (B=8, C=256, 100) -> qf bf16 rows (s*8+b, c)
__global__ void prep_qf(const float* __restrict__ qf4, __bf16* __restrict__ out) {
    int i = blockIdx.x * 256 + threadIdx.x;
    if (i < 800 * 256) {
        int c = i & 255, row = i >> 8;
        int b = row & 7, s = row >> 3;
        out[i] = (__bf16)qf4[((size_t)b * 256 + c) * 100 + s];
    }
}

// ---------------------------------------------------------------------------
// WMMA GEMM:  C[M,N] = A[M,K](bf16) * W[N,K](bf16)^T + bias[N]
// 256 thr = 8 waves; block tile 64(M) x 32(N); K-slab 64; 2 WMMA / wave / slab.
// A buffers are padded to a multiple of 64 rows -> staging is UNGUARDED and
// done with global_load_async_to_lds_b128 into double-buffered LDS.
// Requires: K % 64 == 0, N % 32 == 0 (true for all calls: K=256, N in {256,32}).
// ---------------------------------------------------------------------------
#define BM 64
#define BN 32
#define BK 64

__global__ void gemm_bf16_wmma(const __bf16* __restrict__ A,
                               const __bf16* __restrict__ W,
                               const float*  __restrict__ bias,
                               float*  __restrict__ Cf,   // may be null
                               __bf16* __restrict__ Cb,   // may be null
                               int M, int N, int K, int relu)
{
    __shared__ __bf16 As[2][BM][BK];   // 2 x 8 KB
    __shared__ __bf16 Ws[2][BN][BK];   // 2 x 4 KB
    const int tid  = threadIdx.x;
    const int lane = tid & 31;
    const int w    = tid >> 5;
    const int m0 = blockIdx.x * BM, n0 = blockIdx.y * BN;
    const int mw = (w & 3) * 16,   nw = (w >> 2) * 16;
    const int l15  = lane & 15;
    const int half = lane >> 4;

    // Per-thread async staging: A = 512 16B-chunks (2/thr), W = 256 (1/thr).
    auto stage = [&](int buf, int k0) {
        #pragma unroll
        for (int i = 0; i < 2; ++i) {
            int c = tid + i * 256;
            int row = c >> 3, sub = (c & 7) * 8;           // 8 chunks per 128B row
            async_ld_b128(A + (size_t)(m0 + row) * K + k0 + sub, &As[buf][row][sub]);
        }
        {
            int row = tid >> 3, sub = (tid & 7) * 8;
            async_ld_b128(W + (size_t)(n0 + row) * K + k0 + sub, &Ws[buf][row][sub]);
        }
    };

    v8f acc = {};
    const int niter = K / BK;
    stage(0, 0);
    for (int i = 0; i < niter; ++i) {
        if (i + 1 < niter) {
            stage((i + 1) & 1, (i + 1) * BK);
            // 3 ops of stage(i+1) may stay in flight; in-order completion
            // guarantees stage(i) has fully landed in LDS.
            asm volatile("s_wait_asynccnt 0x3" ::: "memory");
        } else {
            asm volatile("s_wait_asynccnt 0x0" ::: "memory");
        }
        __syncthreads();
        const int buf = i & 1;
        #pragma unroll
        for (int ks = 0; ks < 2; ++ks) {
            v16bf af, bf;
            const __bf16* pa0 = &As[buf][mw + l15][ks * 32 + 16 * half];
            const __bf16* pa1 = pa0 + 8;
            #pragma unroll
            for (int e = 0; e < 8; ++e) { af[e] = pa0[e]; af[8 + e] = pa1[e]; }
            const __bf16* pb = &Ws[buf][nw + l15][ks * 32 + 16 * half];
            #pragma unroll
            for (int e = 0; e < 16; ++e) bf[e] = pb[e];
            acc = wmma_bf16(af, bf, acc);
        }
        __syncthreads();   // protect buf before it is re-staged two slabs later
    }

    // epilogue: lane col n, rows m = j + 8*half
    int n = n0 + nw + l15;
    float bv = (n < N) ? bias[n] : 0.f;
    #pragma unroll
    for (int j = 0; j < 8; ++j) {
        int m = m0 + mw + j + 8 * half;
        if (m < M && n < N) {
            float v = acc[j] + bv;
            if (relu) v = fmaxf(v, 0.f);
            if (Cf) Cf[(size_t)m * N + n] = v;
            if (Cb) Cb[(size_t)m * N + n] = (__bf16)v;
        }
    }
}

// ---------------------------------------------------------------------------
// Self-attention over queries (tiny: S=100, B=8, H=8, dh=32). Scalar f32.
// One block per (b,h). Writes concat-head output as bf16 rows (s*8+b, c).
// ---------------------------------------------------------------------------
__global__ void self_attn(const float* __restrict__ q, const float* __restrict__ k,
                          const float* __restrict__ v, __bf16* __restrict__ outb)
{
    __shared__ float P[100 * 100];   // 40 KB
    const int bh = blockIdx.x, b = bh >> 3, h = bh & 7;
    const int tid = threadIdx.x;
    const float scale = 0.17677669529663687f;   // 1/sqrt(32)
    for (int s = tid; s < 100; s += 128) {
        const float* qp = q + ((size_t)s * 8 + b) * 256 + h * 32;
        float mx = -1e30f;
        for (int t = 0; t < 100; ++t) {
            const float* kp = k + ((size_t)t * 8 + b) * 256 + h * 32;
            float d = 0.f;
            #pragma unroll
            for (int e = 0; e < 32; ++e) d += qp[e] * kp[e];
            d *= scale; P[s * 100 + t] = d; mx = fmaxf(mx, d);
        }
        float sum = 0.f;
        for (int t = 0; t < 100; ++t) { float e = __expf(P[s * 100 + t] - mx); P[s * 100 + t] = e; sum += e; }
        float inv = 1.f / sum;
        float o[32];
        #pragma unroll
        for (int e = 0; e < 32; ++e) o[e] = 0.f;
        for (int t = 0; t < 100; ++t) {
            const float* vp = v + ((size_t)t * 8 + b) * 256 + h * 32;
            float p = P[s * 100 + t] * inv;
            #pragma unroll
            for (int e = 0; e < 32; ++e) o[e] += p * vp[e];
        }
        __bf16* ob = outb + ((size_t)s * 8 + b) * 256 + h * 32;
        #pragma unroll
        for (int e = 0; e < 32; ++e) ob[e] = (__bf16)o[e];
    }
}

// ---------------------------------------------------------------------------
// Fused cross-attention: one block (256 thr / 8 waves) per proposal (b,p).
//   Stage 1: kc = feats @ Wk^T, vc = feats @ Wv^T  (WMMA, results into LDS)
//            Kc[208][256] bf16 (t-major, t padded w/ zeros)
//            VcT[256][224] bf16 (d-major so PV B-operand is contiguous)
//   Stage 2 per head: S = qc.Kc^T (one K=32 WMMA per 16x16 tile),
//            row softmax in Sc[112][224] f32, then O = P.Vc (7 K-steps).
// LDS total = 106496 + 114688 + 100352 = 321536 B  (<= 320 KB WGP LDS)
// ---------------------------------------------------------------------------
#define CA_SMEM 321536

__global__ void cross_attn_fused(const float* __restrict__ feats, // (N,256,196) f32
                                 const float* __restrict__ qc,    // (800,256) f32, row = s*8+b
                                 const __bf16* __restrict__ wk,   // (256,256) bf16 (n,k)
                                 const __bf16* __restrict__ wv,
                                 const float* __restrict__ bk,
                                 const float* __restrict__ bv,
                                 __bf16* __restrict__ out)        // (512*100,256) bf16
{
    extern __shared__ char smem[];
    __bf16* Kc  = (__bf16*)smem;                             // [208][256]
    __bf16* VcT = (__bf16*)(smem + 208 * 256 * 2);           // [256][224]
    float*  Sc  = (float*)(smem + 208 * 256 * 2 + 256 * 224 * 2); // [112][224]

    const int bp = blockIdx.x;           // b*64 + p
    const int b  = bp >> 6;
    const float* fptr = feats + (size_t)bp * 256 * 196;
    const int tid = threadIdx.x, lane = tid & 31, w = tid >> 5;
    const int l15 = lane & 15, half = lane >> 4;
    const float scale = 0.17677669529663687f;

    // zero the padded tails (t in [196,208) of Kc, t in [196,224) of VcT)
    for (int i = tid; i < 12 * 256; i += 256) Kc[(196 + i / 256) * 256 + (i % 256)] = (__bf16)0.f;
    for (int i = tid; i < 256 * 28; i += 256) VcT[(i / 28) * 224 + 196 + (i % 28)] = (__bf16)0.f;

    // ---- Stage 1: K/V projections. 26 jobs = 13 t-tiles x {K,V} ----
    for (int job = w; job < 26; job += 8) {
        const int  isv = job >= 13;
        const int  tt  = isv ? job - 13 : job;
        const __bf16* Wm = isv ? wv : wk;
        const float*  bb = isv ? bv : bk;
        const int t = tt * 16 + l15;

        v16bf afr[8];
        #pragma unroll
        for (int ks = 0; ks < 8; ++ks) {
            v16bf a = {};
            if (t < 196) {
                int c0 = ks * 32 + 16 * half;
                #pragma unroll
                for (int e = 0; e < 8; ++e) {
                    a[e]     = (__bf16)fptr[(size_t)(c0 + e) * 196 + t];
                    a[8 + e] = (__bf16)fptr[(size_t)(c0 + 8 + e) * 196 + t];
                }
            }
            afr[ks] = a;
        }
        for (int nt = 0; nt < 16; ++nt) {
            const int n = nt * 16 + l15;
            v8f acc = {};
            #pragma unroll
            for (int ks = 0; ks < 8; ++ks) {
                const __bf16* p = Wm + (size_t)n * 256 + ks * 32 + 16 * half;
                v16bf bfr;
                #pragma unroll
                for (int e = 0; e < 16; ++e) bfr[e] = p[e];
                acc = wmma_bf16(afr[ks], bfr, acc);
            }
            const float bvv = bb[n];
            #pragma unroll
            for (int j = 0; j < 8; ++j) {
                int tr = tt * 16 + j + 8 * half;
                if (tr < 196) {
                    float vv = acc[j] + bvv;
                    if (isv) VcT[(size_t)n * 224 + tr]  = (__bf16)vv;
                    else     Kc [(size_t)tr * 256 + n]  = (__bf16)vv;
                }
            }
        }
    }
    __syncthreads();

    // ---- Stage 2: per-head attention (waves 0..6 own s-tiles 0..6) ----
    for (int h = 0; h < 8; ++h) {
        if (w < 7) {
            const int s = w * 16 + l15;
            v16bf aq = {};
            if (s < 100) {
                const float* qp = qc + ((size_t)s * 8 + b) * 256 + h * 32;
                #pragma unroll
                for (int e = 0; e < 8; ++e) {
                    aq[e]     = (__bf16)qp[16 * half + e];
                    aq[8 + e] = (__bf16)qp[16 * half + 8 + e];
                }
            }
            for (int tt = 0; tt < 13; ++tt) {
                const int tl = tt * 16 + l15;
                const __bf16* kp = Kc + (size_t)tl * 256 + h * 32 + 16 * half;
                v16bf bk16;
                #pragma unroll
                for (int e = 0; e < 16; ++e) bk16[e] = kp[e];
                v8f acc = {};
                acc = wmma_bf16(aq, bk16, acc);           // dh = 32 = one K-step
                #pragma unroll
                for (int j = 0; j < 8; ++j)
                    Sc[(size_t)(w * 16 + j + 8 * half) * 224 + tt * 16 + l15] = acc[j] * scale;
            }
        }
        __syncthreads();

        if (tid < 112) {   // row softmax over T=196, zero the pad cols
            float* row = Sc + (size_t)tid * 224;
            float mx = -1e30f;
            for (int t = 0; t < 196; ++t) mx = fmaxf(mx, row[t]);
            float sum = 0.f;
            for (int t = 0; t < 196; ++t) { float e = __expf(row[t] - mx); row[t] = e; sum += e; }
            float inv = 1.f / sum;
            for (int t = 0; t < 196; ++t) row[t] *= inv;
            for (int t = 196; t < 224; ++t) row[t] = 0.f;
        }
        __syncthreads();

        if (w < 7) {
            #pragma unroll
            for (int dt = 0; dt < 2; ++dt) {
                const int d = h * 32 + dt * 16;
                v8f acc = {};
                for (int kt = 0; kt < 7; ++kt) {
                    const float* ap = Sc + (size_t)(w * 16 + l15) * 224 + kt * 32;
                    v16bf af;
                    #pragma unroll
                    for (int e = 0; e < 8; ++e) {
                        af[e]     = (__bf16)ap[16 * half + e];
                        af[8 + e] = (__bf16)ap[16 * half + 8 + e];
                    }
                    const __bf16* vp = VcT + (size_t)(d + l15) * 224 + kt * 32 + 16 * half;
                    v16bf bf16v;
                    #pragma unroll
                    for (int e = 0; e < 16; ++e) bf16v[e] = vp[e];
                    acc = wmma_bf16(af, bf16v, acc);
                }
                #pragma unroll
                for (int j = 0; j < 8; ++j) {
                    int s = w * 16 + j + 8 * half;
                    if (s < 100)
                        out[((size_t)bp * 100 + s) * 256 + d + l15] = (__bf16)acc[j];
                }
            }
        }
        __syncthreads();
    }
}

// ---------------------------------------------------------------------------
// Score head: scores[bp] = mean_s sigmoid(oc[bp,s,:].ws + bs)
// ---------------------------------------------------------------------------
__global__ void score_head(const __bf16* __restrict__ oc, const float* __restrict__ wsv,
                           const float* __restrict__ bsv, float* __restrict__ out)
{
    __shared__ float red[256];
    const int bp = blockIdx.x, tid = threadIdx.x;
    float accum = 0.f;
    const float wv = wsv[tid];
    for (int s = 0; s < 100; ++s) {
        const __bf16* row = oc + ((size_t)bp * 100 + s) * 256;
        red[tid] = (float)row[tid] * wv;
        __syncthreads();
        for (int off = 128; off > 0; off >>= 1) {
            if (tid < off) red[tid] += red[tid + off];
            __syncthreads();
        }
        if (tid == 0) accum += 1.f / (1.f + __expf(-(red[0] + bsv[0])));
        __syncthreads();
    }
    if (tid == 0) out[bp] = accum * 0.01f;
}

// h1 mean over S: (512*100,256) bf16 -> (512,256) bf16
__global__ void mean_s(const __bf16* __restrict__ h1, __bf16* __restrict__ outb)
{
    const int bp = blockIdx.x, c = threadIdx.x;
    float s = 0.f;
    for (int i = 0; i < 100; ++i) s += (float)h1[((size_t)bp * 100 + i) * 256 + c];
    outb[(size_t)bp * 256 + c] = (__bf16)(s * 0.01f);
}

// ---------------------------------------------------------------------------
// Host-side orchestration
// ---------------------------------------------------------------------------
extern "C" void kernel_launch(void* const* d_in, const int* in_sizes, int n_in,
                              void* d_out, int out_size, void* d_ws, size_t ws_size,
                              hipStream_t stream) {
    (void)in_sizes; (void)n_in; (void)out_size; (void)ws_size;
    const float* features = (const float*)d_in[0];
    const float* query_features = (const float*)d_in[1];
    const float* wq_s = (const float*)d_in[2];
    const float* wk_s = (const float*)d_in[3];
    const float* wv_s = (const float*)d_in[4];
    const float* bq_s = (const float*)d_in[5];
    const float* bk_s = (const float*)d_in[6];
    const float* bv_s = (const float*)d_in[7];
    const float* wo_s = (const float*)d_in[8];
    const float* bo_s = (const float*)d_in[9];
    const float* wq_c = (const float*)d_in[10];
    const float* wk_c = (const float*)d_in[11];
    const float* wv_c = (const float*)d_in[12];
    const float* bq_c = (const float*)d_in[13];
    const float* bk_c = (const float*)d_in[14];
    const float* bv_c = (const float*)d_in[15];
    const float* wo_c = (const float*)d_in[16];
    const float* bo_c = (const float*)d_in[17];
    const float* w1   = (const float*)d_in[18];
    const float* b1   = (const float*)d_in[19];
    const float* w2   = (const float*)d_in[20];
    const float* b2   = (const float*)d_in[21];
    const float* wsv  = (const float*)d_in[22];
    const float* bsv  = (const float*)d_in[23];
    float* outp = (float*)d_out;

    char* base = (char*)d_ws;
    size_t off = 0;
    auto alloc = [&](size_t bytes) -> void* {
        void* p = (void*)(base + off);
        off += (bytes + 255) & ~(size_t)255;
        return p;
    };
    // NOTE: all GEMM A-operands padded to a multiple of 64 rows (unguarded
    // async staging reads up to the padded row count).
    __bf16* wq_s_b = (__bf16*)alloc(65536 * 2);
    __bf16* wk_s_b = (__bf16*)alloc(65536 * 2);
    __bf16* wv_s_b = (__bf16*)alloc(65536 * 2);
    __bf16* wo_s_b = (__bf16*)alloc(65536 * 2);
    __bf16* wq_c_b = (__bf16*)alloc(65536 * 2);
    __bf16* wk_c_b = (__bf16*)alloc(65536 * 2);
    __bf16* wv_c_b = (__bf16*)alloc(65536 * 2);
    __bf16* wo_c_b = (__bf16*)alloc(65536 * 2);
    __bf16* w1_b   = (__bf16*)alloc(65536 * 2);
    __bf16* w2_b   = (__bf16*)alloc(8192 * 2);
    __bf16* qf_b   = (__bf16*)alloc((size_t)832 * 256 * 2);   // 800 used
    float*  q_f    = (float*)alloc((size_t)800 * 256 * 4);
    float*  k_f    = (float*)alloc((size_t)800 * 256 * 4);
    float*  v_f    = (float*)alloc((size_t)800 * 256 * 4);
    __bf16* atts_b = (__bf16*)alloc((size_t)832 * 256 * 2);   // 800 used
    __bf16* qsa_b  = (__bf16*)alloc((size_t)832 * 256 * 2);   // 800 used
    float*  qc_f   = (float*)alloc((size_t)800 * 256 * 4);
    __bf16* attc_b = (__bf16*)alloc((size_t)51200 * 256 * 2); // reused as h1_b
    __bf16* oc_b   = (__bf16*)alloc((size_t)51200 * 256 * 2);
    __bf16* h1m_b  = (__bf16*)alloc((size_t)512 * 256 * 2);
    __bf16* h1_b   = attc_b;  // attc dead once oc is produced

    // 1) weight prep (f32 -> bf16, torch (out,in) layout kept as-is)
    cvt_f32_bf16<<<256, 256, 0, stream>>>(wq_s, wq_s_b, 65536);
    cvt_f32_bf16<<<256, 256, 0, stream>>>(wk_s, wk_s_b, 65536);
    cvt_f32_bf16<<<256, 256, 0, stream>>>(wv_s, wv_s_b, 65536);
    cvt_f32_bf16<<<256, 256, 0, stream>>>(wo_s, wo_s_b, 65536);
    cvt_f32_bf16<<<256, 256, 0, stream>>>(wq_c, wq_c_b, 65536);
    cvt_f32_bf16<<<256, 256, 0, stream>>>(wk_c, wk_c_b, 65536);
    cvt_f32_bf16<<<256, 256, 0, stream>>>(wv_c, wv_c_b, 65536);
    cvt_f32_bf16<<<256, 256, 0, stream>>>(wo_c, wo_c_b, 65536);
    cvt_f32_bf16<<<256, 256, 0, stream>>>(w1,   w1_b,   65536);
    cvt_f32_bf16<<<32,  256, 0, stream>>>(w2,   w2_b,   8192);
    prep_qf<<<800, 256, 0, stream>>>(query_features, qf_b);

    // 2) self-attention Q/K/V projections (M=800)
    dim3 g1(13, 8);
    gemm_bf16_wmma<<<g1, 256, 0, stream>>>(qf_b, wq_s_b, bq_s, q_f, nullptr, 800, 256, 256, 0);
    gemm_bf16_wmma<<<g1, 256, 0, stream>>>(qf_b, wk_s_b, bk_s, k_f, nullptr, 800, 256, 256, 0);
    gemm_bf16_wmma<<<g1, 256, 0, stream>>>(qf_b, wv_s_b, bv_s, v_f, nullptr, 800, 256, 256, 0);
    self_attn<<<64, 128, 0, stream>>>(q_f, k_f, v_f, atts_b);
    gemm_bf16_wmma<<<g1, 256, 0, stream>>>(atts_b, wo_s_b, bo_s, nullptr, qsa_b, 800, 256, 256, 0);
    gemm_bf16_wmma<<<g1, 256, 0, stream>>>(qsa_b, wq_c_b, bq_c, qc_f, nullptr, 800, 256, 256, 0);

    // 3) fused cross-attention (one WG per proposal, 320KB-class LDS)
    cross_attn_fused<<<512, 256, CA_SMEM, stream>>>(features, qc_f, wk_c_b, wv_c_b,
                                                    bk_c, bv_c, attc_b);

    // 4) output projection + heads (M=51200)
    dim3 g2(800, 8);
    gemm_bf16_wmma<<<g2, 256, 0, stream>>>(attc_b, wo_c_b, bo_c, nullptr, oc_b, 51200, 256, 256, 0);
    score_head<<<512, 256, 0, stream>>>(oc_b, wsv, bsv, outp + 512 * 32);
    gemm_bf16_wmma<<<g2, 256, 0, stream>>>(oc_b, w1_b, b1, nullptr, h1_b, 51200, 256, 256, 1);
    mean_s<<<512, 256, 0, stream>>>(h1_b, h1m_b);
    dim3 g3(8, 1);
    gemm_bf16_wmma<<<g3, 256, 0, stream>>>(h1m_b, w2_b, b2, outp, nullptr, 512, 32, 256, 0);
}
